// DictPlenoxels_53781580481032
// MI455X (gfx1250) — matrix operand order repaired and checked by reading, not compiled, not run on hardware
//
#include <hip/hip_runtime.h>
#include <hip/hip_bf16.h>
#include <math.h>

typedef __attribute__((ext_vector_type(16))) _Float16 v16h;
typedef __attribute__((ext_vector_type(8)))  float    v8f;

// ---- problem constants (match reference) ----
#define RADIUS_F   1.3f
#define COARSE_I   64
#define NSAMP      767                      // N_INTRS - 1
#define NTILES     48                       // ceil(768/16)
#define STEP_F     (RADIUS_F * 2.0f / 64.0f / 2.0f / 2.0f)   // 0.01015625
#define DATA_DIM_I 28

__device__ __forceinline__ int iclamp(int v, int lo, int hi) {
    return v < lo ? lo : (v > hi ? hi : v);
}

__global__ __launch_bounds__(128)
void DictPlenoxels_wmma_kernel(const float* __restrict__ rays_o,
                               const float* __restrict__ rays_d,
                               const float* __restrict__ grid,
                               const float* __restrict__ atoms,
                               float* __restrict__ out, int B) {
    // LDS: atoms staged as WMMA-B fragments (f16), per-wave D spill, ray buffers
    __shared__ v16h  sBfrag[1024];       // 2 chunks * 16 ksteps * 32 lanes, 32 KB
    __shared__ float sOut[4 * 16 * 32];  // per-wave 16x32 D tile, 8 KB
    __shared__ float sAlpha[768];
    __shared__ float sRgb[768 * 3];

    const int tid  = threadIdx.x;
    const int lane = tid & 31;
    const int wave = tid >> 5;
    const int m    = lane & 15;          // sample row within tile
    const int g    = (lane >> 4) & 1;    // wave-half: selects K+8 / a-bit3 per ISA A-layout
    const int ray  = blockIdx.x;

    // ---------- Phase A: stage atoms -> f16 WMMA-B fragment layout ----------
    // B 32x16 per k-step: lanes 0-15 hold K 0-7,16-23 (VGPR j half h -> K=2j+h+8*(j>>2)),
    // lanes 16-31 hold K 8-15,24-31 (same +8); N = lane&15.  Fragment e = 2j+h.
    {
        _Float16* bf = (_Float16*)sBfrag;
        for (int idx = tid; idx < 16384; idx += 128) {
            int e     = idx & 15;
            int ln    = (idx >> 4) & 31;
            int kk    = (idx >> 9) & 15;
            int chunk = (idx >> 13) & 1;
            int kr  = ((ln >> 4) << 3) + (e & 7) + ((e >> 3) << 4);       // 0..31 local K
            int k   = kk * 32 + kr;               // 0..511 = f*64 + a
            int col = chunk * 16 + (ln & 15);     // 0..31 output column d
            int f = k >> 6, a = k & 63;
            float v = (col < DATA_DIM_I) ? atoms[(f * 64 + a) * DATA_DIM_I + col] : 0.0f;
            bf[idx] = (_Float16)v;
        }
    }

    // ---------- per-ray setup (every thread, registers) ----------
    const float o0 = rays_o[ray * 3 + 0], o1 = rays_o[ray * 3 + 1], o2 = rays_o[ray * 3 + 2];
    const float d0 = rays_d[ray * 3 + 0], d1 = rays_d[ray * 3 + 1], d2 = rays_d[ray * 3 + 2];
    const float dnorm = sqrtf(d0 * d0 + d1 * d1 + d2 * d2);
    float start = -3.4e38f;
    {
        float p, n;
        p = ( RADIUS_F - o0) / d0; n = (-RADIUS_F - o0) / d0; start = fmaxf(start, fminf(p, n));
        p = ( RADIUS_F - o1) / d1; n = (-RADIUS_F - o1) / d1; start = fmaxf(start, fminf(p, n));
        p = ( RADIUS_F - o2) / d2; n = (-RADIUS_F - o2) / d2; start = fmaxf(start, fminf(p, n));
    }
    const float dist = STEP_F * dnorm;

    // SH basis of the (normalized) ray direction
    float shm[9];
    {
        const float inv = 1.0f / dnorm;
        const float x = d0 * inv, y = d1 * inv, z = d2 * inv;
        shm[0] =  0.28209479177387814f;
        shm[1] = -0.4886025119029199f * y;
        shm[2] =  0.4886025119029199f * z;
        shm[3] = -0.4886025119029199f * x;
        shm[4] =  1.0925484305920792f * x * y;
        shm[5] = -1.0925484305920792f * y * z;
        shm[6] =  0.31539156525252005f * (2.0f * z * z - x * x - y * y);
        shm[7] = -1.0925484305920792f * x * z;
        shm[8] =  0.5462742152960396f * (x * x - y * y);
    }

    __syncthreads();

    // ---------- Phase B: per-16-sample tiles, D(16x32) = X(16x512) * W(512x32) ----------
    for (int tile = wave; tile < NTILES; tile += 4) {
        const int  s     = tile * 16 + m;
        const bool valid = s < NSAMP;
        const int  sc    = valid ? s : (NSAMP - 1);
        const float t_r  = start + (float)sc * STEP_F;

        const float px = o0 + t_r * d0;
        const float py = o1 + t_r * d1;
        const float pz = o2 + t_r * d2;
        const bool inside = (px > -RADIUS_F) && (px < RADIUS_F) &&
                            (py > -RADIUS_F) && (py < RADIUS_F) &&
                            (pz > -RADIUS_F) && (pz < RADIUS_F);
        const bool msk = valid && inside;

        const float qx = fminf(fmaxf((px + RADIUS_F) * (0.5f / RADIUS_F), 0.0f), 1.0f - 1e-6f);
        const float qy = fminf(fmaxf((py + RADIUS_F) * (0.5f / RADIUS_F), 0.0f), 1.0f - 1e-6f);
        const float qz = fminf(fmaxf((pz + RADIUS_F) * (0.5f / RADIUS_F), 0.0f), 1.0f - 1e-6f);

        const float pcx = qx * 64.0f, pcy = qy * 64.0f, pcz = qz * 64.0f;
        const int cx = iclamp((int)floorf(pcx), 0, 63);
        const int cy = iclamp((int)floorf(pcy), 0, 63);
        const int cz = iclamp((int)floorf(pcz), 0, 63);
        const float lx = pcx - (float)cx, ly = pcy - (float)cy, lz = pcz - (float)cz;
        const int clin = (cx * 64 + cy) * 64 + cz;

        // trilinear fine weights folded into the 8 FINE^3 slots
        float we[8];
        {
            const float ux = lx * 2.0f - 0.5f, uy = ly * 2.0f - 0.5f, uz = lz * 2.0f - 0.5f;
            const float fx = floorf(ux), fy = floorf(uy), fz = floorf(uz);
            const float tx = ux - fx, ty = uy - fy, tz = uz - fz;
            const int ix0 = iclamp((int)fx, 0, 1),     iy0 = iclamp((int)fy, 0, 1),     iz0 = iclamp((int)fz, 0, 1);
            const int ix1 = iclamp((int)fx + 1, 0, 1), iy1 = iclamp((int)fy + 1, 0, 1), iz1 = iclamp((int)fz + 1, 0, 1);
            const float wxs0 = (ix0 == 0 ? 1.0f - tx : 0.0f) + (ix1 == 0 ? tx : 0.0f);
            const float wxs1 = (ix0 == 1 ? 1.0f - tx : 0.0f) + (ix1 == 1 ? tx : 0.0f);
            const float wys0 = (iy0 == 0 ? 1.0f - ty : 0.0f) + (iy1 == 0 ? ty : 0.0f);
            const float wys1 = (iy0 == 1 ? 1.0f - ty : 0.0f) + (iy1 == 1 ? ty : 0.0f);
            const float wzs0 = (iz0 == 0 ? 1.0f - tz : 0.0f) + (iz1 == 0 ? tz : 0.0f);
            const float wzs1 = (iz0 == 1 ? 1.0f - tz : 0.0f) + (iz1 == 1 ? tz : 0.0f);
            we[0] = wxs0 * wys0 * wzs0;  we[1] = wxs0 * wys0 * wzs1;
            we[2] = wxs0 * wys1 * wzs0;  we[3] = wxs0 * wys1 * wzs1;
            we[4] = wxs1 * wys0 * wzs0;  we[5] = wxs1 * wys0 * wzs1;
            we[6] = wxs1 * wys1 * wzs0;  we[7] = wxs1 * wys1 * wzs1;
        }

        // gather this lane's half of the 64-float grid row (a with bit3 == g)
        float cfl[32];
        {
            const float4* grow = (const float4*)(grid + (size_t)clin * 64);
            #pragma unroll
            for (int gi = 0; gi < 4; ++gi) {
                #pragma unroll
                for (int hf = 0; hf < 2; ++hf) {
                    float4 v4 = grow[gi * 4 + g * 2 + hf];
                    cfl[gi * 8 + hf * 4 + 0] = v4.x;
                    cfl[gi * 8 + hf * 4 + 1] = v4.y;
                    cfl[gi * 8 + hf * 4 + 2] = v4.z;
                    cfl[gi * 8 + hf * 4 + 3] = v4.w;
                }
            }
        }

        // 16 K-steps of v_wmma_f32_16x16x32_f16, two N-chunks sharing each A fragment
        v8f acc0 = {0.f, 0.f, 0.f, 0.f, 0.f, 0.f, 0.f, 0.f};
        v8f acc1 = {0.f, 0.f, 0.f, 0.f, 0.f, 0.f, 0.f, 0.f};
        #pragma unroll
        for (int kk = 0; kk < 16; ++kk) {
            v16h af;
            #pragma unroll
            for (int i = 0; i < 8; ++i) {
                #pragma unroll
                for (int h = 0; h < 2; ++h) {
                    // q = kk*32 + (i&3)*2 + (i>>2)*16 + g*8 + h ; bit3 (= g) does not
                    // feed f = q>>6 nor cfl index, so fold it out -> constant indices
                    const int q0i = kk * 32 + ((i & 3) << 1) + ((i >> 2) << 4) + h;
                    const int f   = q0i >> 6;
                    const int j   = (((q0i >> 4) & 3) << 3) | (q0i & 7);
                    af[i * 2 + h] = (_Float16)(we[f] * cfl[j]);
                }
            }
            v16h bf0 = sBfrag[(0 * 16 + kk) * 32 + lane];
            v16h bf1 = sBfrag[(1 * 16 + kk) * 32 + lane];
            acc0 = __builtin_amdgcn_wmma_f32_16x16x32_f16(false, af, false, bf0, (short)0, acc0, false, false);
            acc1 = __builtin_amdgcn_wmma_f32_16x16x32_f16(false, af, false, bf1, (short)0, acc1, false, false);
        }

        // spill D through LDS: lane holds column n=m, rows v+8g
        #pragma unroll
        for (int v = 0; v < 8; ++v) {
            sOut[wave * 512 + (g * 8 + v) * 32 +  0 + m] = acc0[v];
            sOut[wave * 512 + (g * 8 + v) * 32 + 16 + m] = acc1[v];
        }
        __syncthreads();

        // per-sample post-processing: sigma/alpha + SH shading (row per lane)
        if (lane < 16 && valid) {
            const float* orow = &sOut[wave * 512 + lane * 32];
            const float sig = msk ? fmaxf(orow[27], 0.0f) : 0.0f;
            sAlpha[s] = 1.0f - expf(-sig * dist);
            #pragma unroll
            for (int c = 0; c < 3; ++c) {
                float pre = 0.0f;
                #pragma unroll
                for (int k = 0; k < 9; ++k) pre += orow[c * 9 + k] * shm[k];
                pre = msk ? pre : 0.0f;
                sRgb[s * 3 + c] = 1.0f / (1.0f + expf(-pre));
            }
        }
        __syncthreads();
    }

    // ---------- Phase C: serial transmittance scan + composite ----------
    if (tid == 0) {
        float T = 1.0f, c0 = 0.f, c1 = 0.f, c2 = 0.f, aw = 0.f, dep = 0.f;
        for (int s = 0; s < NSAMP; ++s) {
            const float a = sAlpha[s];
            const float w = a * T;
            c0 += w * sRgb[s * 3 + 0];
            c1 += w * sRgb[s * 3 + 1];
            c2 += w * sRgb[s * 3 + 2];
            aw += w;
            dep += w * (start + (float)s * STEP_F);
            T *= (1.0f - a + 1e-10f);
        }
        float* orgb = out + (size_t)ray * 3;
        orgb[0] = c0 + (1.0f - aw);
        orgb[1] = c1 + (1.0f - aw);
        orgb[2] = c2 + (1.0f - aw);
        out[(size_t)B * 3 + (size_t)B * NSAMP + ray] = dep;   // depth
    }
    __syncthreads();
    {
        float* oal = out + (size_t)B * 3 + (size_t)ray * NSAMP;
        for (int sidx = tid; sidx < NSAMP; sidx += 128) oal[sidx] = sAlpha[sidx];
    }
}

extern "C" void kernel_launch(void* const* d_in, const int* in_sizes, int n_in,
                              void* d_out, int out_size, void* d_ws, size_t ws_size,
                              hipStream_t stream) {
    const float* rays_o = (const float*)d_in[0];
    const float* rays_d = (const float*)d_in[1];
    const float* grid   = (const float*)d_in[2];
    const float* atoms  = (const float*)d_in[3];
    // d_in[4] = grid_id (unused scalar)
    float* out = (float*)d_out;
    const int B = in_sizes[0] / 3;

    DictPlenoxels_wmma_kernel<<<B, 128, 0, stream>>>(rays_o, rays_d, grid, atoms, out, B);
}
